// CrossAttention_73186242724243
// MI455X (gfx1250) — matrix operand order repaired and checked by reading, not compiled, not run on hardware
//
#include <hip/hip_runtime.h>
#include <hip/hip_bf16.h>
#include <stdint.h>

typedef __attribute__((ext_vector_type(16))) __bf16 v16bf;
typedef __attribute__((ext_vector_type(8)))  float  v8f;
typedef __attribute__((ext_vector_type(2)))  float  v2f;
typedef __attribute__((ext_vector_type(4)))  __bf16 v4bf;

__device__ __forceinline__ v8f zero8() {
    v8f v;
#pragma unroll
    for (int i = 0; i < 8; ++i) v[i] = 0.f;
    return v;
}

// ---------------------------------------------------------------------------
// fp32 -> bf16 weight pre-conversion (row-major preserved)
// ---------------------------------------------------------------------------
__global__ __launch_bounds__(256) void cvt_f32_bf16(
    const float* __restrict__ src, __bf16* __restrict__ dst, int n)
{
    int i = blockIdx.x * 256 + threadIdx.x;
    if (i < n) dst[i] = (__bf16)src[i];
}

// ---------------------------------------------------------------------------
// GEMM: C[M,320] = (A[M,K] @ W[320,K]^T + bias) * gamma
// W pre-converted to bf16. block = 128 threads (4 waves), 64 M-rows / block.
// Wave computes 4 M-tiles x 5 N-tiles (tiles {w, w+4, ..., w+16}) = 320 cols.
// ---------------------------------------------------------------------------
template <int K>
__global__ __launch_bounds__(128) void gemm_bf16_k(
    const float* __restrict__ A, const __bf16* __restrict__ W,
    const float* __restrict__ gamma, const float* __restrict__ bias,
    float* __restrict__ C, int M)
{
    constexpr int SA = K + 8;  // padded LDS row stride (bf16) -> conflict-free
    __shared__ __align__(16) __bf16 As[64 * SA];

    const int tid  = threadIdx.x;
    const int wave = tid >> 5;
    const int lane = tid & 31;
    const int l16  = lane & 15;
    const int half = lane >> 4;
    const size_t m0 = (size_t)blockIdx.x * 64;

    // stage A panel (64 x K) into LDS as bf16; OOB rows -> 0
    constexpr int K4 = K / 4;
    for (int idx = tid; idx < 64 * K4; idx += 128) {
        int r = idx / K4, c4 = idx - r * K4;
        float4 a4 = make_float4(0.f, 0.f, 0.f, 0.f);
        if (m0 + r < (size_t)M)
            a4 = *(const float4*)&A[(m0 + r) * K + c4 * 4];
        v4bf p;
        p.x = (__bf16)a4.x; p.y = (__bf16)a4.y;
        p.z = (__bf16)a4.z; p.w = (__bf16)a4.w;
        *(v4bf*)&As[r * SA + c4 * 4] = p;
    }
    __syncthreads();

    v8f acc[5][4];
#pragma unroll
    for (int t = 0; t < 5; ++t)
#pragma unroll
        for (int mt = 0; mt < 4; ++mt) acc[t][mt] = zero8();

    for (int k0 = 0; k0 < K; k0 += 32) {
        // A fragments: lane holds row M = mt*16 + l16;
        // elems 0..7 = K(k0+half*8..+7), elems 8..15 = K(k0+16+half*8..+7)
        union { v16bf v; uint4 q[2]; } af[4];
#pragma unroll
        for (int mt = 0; mt < 4; ++mt) {
            const __bf16* ap = &As[(mt * 16 + l16) * SA + k0 + half * 8];
            af[mt].q[0] = *(const uint4*)(ap);
            af[mt].q[1] = *(const uint4*)(ap + 16);
        }
#pragma unroll
        for (int t = 0; t < 5; ++t) {
            const int n0 = (wave + 4 * t) * 16;
            // B fragment (pre-converted bf16): elems 0..7 = K(k0+half*8..+7), elems 8..15 = K(k0+16+half*8..+7)
            const __bf16* wp = &W[(size_t)(n0 + l16) * K + k0 + half * 8];
            union { v16bf v; uint4 q[2]; } bf;
            bf.q[0] = *(const uint4*)(wp);
            bf.q[1] = *(const uint4*)(wp + 16);
#pragma unroll
            for (int mt = 0; mt < 4; ++mt) {
                acc[t][mt] = __builtin_amdgcn_wmma_f32_16x16x32_bf16(
                    false, af[mt].v, false, bf.v, (short)0, acc[t][mt],
                    false, false);
            }
        }
    }

    // epilogue: C layout row = mt*16 + half*8 + r, col = n0 + l16
#pragma unroll
    for (int t = 0; t < 5; ++t) {
        const int col = (wave + 4 * t) * 16 + l16;
        const float g = gamma[col];
        const float b = bias ? bias[col] : 0.f;
#pragma unroll
        for (int mt = 0; mt < 4; ++mt) {
#pragma unroll
            for (int r = 0; r < 8; ++r) {
                const size_t row = m0 + mt * 16 + half * 8 + r;
                if (row < (size_t)M)
                    C[row * 320 + col] = (acc[t][mt][r] + b) * g;
            }
        }
    }
}

// ---------------------------------------------------------------------------
// Attention core: one block per (b, h, 128 queries). 8 waves, 16 q each.
// scores (16x80) and P@V (16x40) via exact f32 WMMA 16x16x4.
// O written in-place over Q (block reads exactly the region it writes).
// ---------------------------------------------------------------------------
__global__ __launch_bounds__(256) void attention_f32(
    const float* __restrict__ Q, const float* __restrict__ Kb,
    const float* __restrict__ Vb, float* __restrict__ O)
{
    constexpr int KSTR = 44;  // padded row stride (floats) for K/V tiles
    constexpr int PSTR = 84;  // padded row stride for probability staging
    __shared__ __align__(16) float Kh[80 * KSTR];
    __shared__ __align__(16) float Vh[80 * KSTR];
    __shared__ __align__(16) float Ps[8 * 16 * PSTR];

    const int tid  = threadIdx.x;
    const int wave = tid >> 5;
    const int lane = tid & 31;
    const int l16  = lane & 15;
    const int half = lane >> 4;
    const int h    = blockIdx.y;
    const int b    = blockIdx.z;
    const int q0   = blockIdx.x * 128 + wave * 16;

    // stage K,V head slices (77x40 -> 80x40, zero padded)
    for (int idx = tid; idx < 80 * 40; idx += 256) {
        int key = idx / 40, d = idx - key * 40;
        float kv = 0.f, vv = 0.f;
        if (key < 77) {
            size_t src = ((size_t)b * 77 + key) * 320 + h * 40 + d;
            kv = Kb[src];
            vv = Vb[src];
        }
        Kh[key * KSTR + d] = kv;
        Vh[key * KSTR + d] = vv;
    }
    __syncthreads();

    // ---- scores = Q @ K^T : 5 key-tiles, K-dim 40 in steps of 4 (f32 WMMA)
    v8f sc[5];
#pragma unroll
    for (int t = 0; t < 5; ++t) sc[t] = zero8();

    const float* qbase = Q + ((size_t)b * 4096 + q0 + l16) * 320 + h * 40;
#pragma unroll
    for (int k = 0; k < 40; k += 4) {
        // A frag: row M=l16, elem j = K(k+half*2+j)
        v2f a = *(const v2f*)(qbase + k + half * 2);
#pragma unroll
        for (int t = 0; t < 5; ++t) {
            // B frag: col = key (t*16+l16), elem j = Kh[key][k+half*2+j]
            v2f bb = *(const v2f*)&Kh[(t * 16 + l16) * KSTR + k + half * 2];
            sc[t] = __builtin_amdgcn_wmma_f32_16x16x4_f32(
                false, a, false, bb, (short)0, sc[t], false, false);
        }
    }

    // scale + mask padded keys
    const float scale = 0.15811388300841897f;  // 40^-0.5
#pragma unroll
    for (int t = 0; t < 5; ++t) {
        const int key = t * 16 + l16;
#pragma unroll
        for (int r = 0; r < 8; ++r)
            sc[t][r] = (key < 77) ? sc[t][r] * scale : -1e30f;
    }

    // softmax: row (half*8+r) lives in 16 lanes of this half x 5 tiles
#pragma unroll
    for (int r = 0; r < 8; ++r) {
        float mx = sc[0][r];
#pragma unroll
        for (int t = 1; t < 5; ++t) mx = fmaxf(mx, sc[t][r]);
        for (int off = 1; off < 16; off <<= 1) mx = fmaxf(mx, __shfl_xor(mx, off));
        float s = 0.f;
#pragma unroll
        for (int t = 0; t < 5; ++t) {
            float e = __expf(sc[t][r] - mx);
            sc[t][r] = e;
            s += e;
        }
        for (int off = 1; off < 16; off <<= 1) s += __shfl_xor(s, off);
        const float inv = 1.f / s;
#pragma unroll
        for (int t = 0; t < 5; ++t) sc[t][r] *= inv;
    }

    // re-stage P (C-layout -> row-major) through this wave's LDS region
    float* pw = &Ps[wave * 16 * PSTR];
#pragma unroll
    for (int t = 0; t < 5; ++t)
#pragma unroll
        for (int r = 0; r < 8; ++r)
            pw[(half * 8 + r) * PSTR + t * 16 + l16] = sc[t][r];
    // same-wave LDS ops complete in order; no barrier needed

    // ---- O = P @ V : 3 d-tiles (40 cols, clamp pads), K-dim 80 step 4
    v8f oc[3];
#pragma unroll
    for (int t = 0; t < 3; ++t) oc[t] = zero8();

#pragma unroll
    for (int kk = 0; kk < 80; kk += 4) {
        v2f a = *(const v2f*)&pw[l16 * PSTR + kk + half * 2];
#pragma unroll
        for (int t = 0; t < 3; ++t) {
            const int d = min(t * 16 + l16, 39);  // pad cols duplicate, dropped
            v2f bb;
            bb.x = Vh[(kk + half * 2 + 0) * KSTR + d];
            bb.y = Vh[(kk + half * 2 + 1) * KSTR + d];
            oc[t] = __builtin_amdgcn_wmma_f32_16x16x4_f32(
                false, a, false, bb, (short)0, oc[t], false, false);
        }
    }

#pragma unroll
    for (int t = 0; t < 3; ++t) {
        const int d = t * 16 + l16;
        if (d < 40) {
#pragma unroll
            for (int r = 0; r < 8; ++r)
                O[((size_t)b * 4096 + q0 + half * 8 + r) * 320 + h * 40 + d] =
                    oc[t][r];
        }
    }
}

// ---------------------------------------------------------------------------
extern "C" void kernel_launch(void* const* d_in, const int* in_sizes, int n_in,
                              void* d_out, int out_size, void* d_ws,
                              size_t ws_size, hipStream_t stream)
{
    (void)in_sizes; (void)n_in; (void)out_size; (void)ws_size;
    const float* x   = (const float*)d_in[0];   // [32,4096,320]
    const float* ctx = (const float*)d_in[1];   // [32,77,768]
    const float* Wq  = (const float*)d_in[2];   // [320,320]
    const float* Wk  = (const float*)d_in[3];   // [320,768]
    const float* Wv  = (const float*)d_in[4];   // [320,768]
    const float* Wo  = (const float*)d_in[5];   // [320,320]
    const float* bo  = (const float*)d_in[6];   // [320]
    const float* gq  = (const float*)d_in[7];
    const float* gk  = (const float*)d_in[8];
    const float* gv  = (const float*)d_in[9];
    const float* go  = (const float*)d_in[10];

    // workspace layout (fp32 unless noted)
    float* Qbuf = (float*)d_ws;                        // 131072*320 (O in-place)
    float* Kbuf = Qbuf + (size_t)131072 * 320;         // 2464*320
    float* Vbuf = Kbuf + (size_t)2464 * 320;           // 2464*320
    __bf16* Wqb = (__bf16*)(Vbuf + (size_t)2464 * 320);
    __bf16* Wkb = Wqb + (size_t)320 * 320;
    __bf16* Wvb = Wkb + (size_t)320 * 768;
    __bf16* Wob = Wvb + (size_t)320 * 768;

    // pre-convert weights to bf16
    cvt_f32_bf16<<<dim3((320 * 320 + 255) / 256), dim3(256), 0, stream>>>(Wq, Wqb, 320 * 320);
    cvt_f32_bf16<<<dim3((320 * 768 + 255) / 256), dim3(256), 0, stream>>>(Wk, Wkb, 320 * 768);
    cvt_f32_bf16<<<dim3((320 * 768 + 255) / 256), dim3(256), 0, stream>>>(Wv, Wvb, 320 * 768);
    cvt_f32_bf16<<<dim3((320 * 320 + 255) / 256), dim3(256), 0, stream>>>(Wo, Wob, 320 * 320);

    const dim3 blk(128);
    gemm_bf16_k<320><<<dim3(131072 / 64), blk, 0, stream>>>(x,   Wqb, gq, nullptr, Qbuf, 131072);
    gemm_bf16_k<768><<<dim3((2464 + 63) / 64), blk, 0, stream>>>(ctx, Wkb, gk, nullptr, Kbuf, 2464);
    gemm_bf16_k<768><<<dim3((2464 + 63) / 64), blk, 0, stream>>>(ctx, Wvb, gv, nullptr, Vbuf, 2464);
    attention_f32<<<dim3(32, 8, 32), dim3(256), 0, stream>>>(Qbuf, Kbuf, Vbuf, Qbuf);
    gemm_bf16_k<320><<<dim3(131072 / 64), blk, 0, stream>>>(Qbuf, Wob, go, bo,
                                                            (float*)d_out, 131072);
}
